// RadialBasis_49366354100598
// MI455X (gfx1250) — matrix operand order repaired and checked by reading, not compiled, not run on hardware
//
#include <hip/hip_runtime.h>

// CDNA5 / gfx1250: wave32, WMMA 16x16x32 f16 -> f32 accum.
typedef _Float16 v16h __attribute__((ext_vector_type(16)));
typedef float    v8f  __attribute__((ext_vector_type(8)));

#define WAVES            16
#define BLOCK            (WAVES * 32)
#define EDGES_PER_BLOCK  (WAVES * 16)
// species(4) * layers(4) * m-tiles(2) * lanes(32) * elems(16) f16 = 32 KB
#define WLDS_HALVES      (4 * 4 * 2 * 32 * 16)

__device__ __forceinline__ v8f wmma16x16x32(v16h a, v16h b, v8f c) {
  // (neg_a, A, neg_b, B, c_mod, C, reuse_a, reuse_b)
  return __builtin_amdgcn_wmma_f32_16x16x32_f16(false, a, false, b, (short)0, c,
                                                false, false);
}

// Fast SiLU: x * rcp(1 + exp(-x)).  v_rcp_f32 (~1 ulp) instead of the IEEE
// divide expansion (div_scale/div_fmas/div_fixup) -- error is far below the
// f16 quantization of the WMMA operands, and exp/rcp run on the TRANS pipe
// which co-executes with the matrix pipe.
__device__ __forceinline__ float silu(float x) {
  return x * __builtin_amdgcn_rcpf(1.0f + __expf(-x));
}

// D tiles (d0: out rows 0..15, d1: rows 16..31; lane = edge column, hl = lane/16,
// reg r holds M = r + 8*hl (+16 for d1)) -> SiLU -> next-layer B fragment
// (lane holds K = e + 16*hl of its edge column). Needs a half-wave swap.
__device__ __forceinline__ v16h silu_transpose(const v8f& d0, const v8f& d1, int hl) {
  float s0[8], s1[8];
#pragma unroll
  for (int e = 0; e < 8; ++e) {
    s0[e] = silu(d0[e]);
    s1[e] = silu(d1[e]);
  }
  float p0[8], p1[8];
#pragma unroll
  for (int e = 0; e < 8; ++e) {
    p0[e] = __shfl_xor(s0[e], 16, 32);
    p1[e] = __shfl_xor(s1[e], 16, 32);
  }
  v16h b;
#pragma unroll
  for (int e = 0; e < 8; ++e) {
    b[e]     = (_Float16)(hl ? p1[e] : s0[e]);   // K = e      (+16 if hl)
    b[e + 8] = (_Float16)(hl ? s1[e] : p0[e]);   // K = e + 8  (+16 if hl)
  }
  return b;
}

struct Params {
  const float* r;
  const int*   species;
  const float* coeffs;       // [600][4][56]
  const float* W1[4];        // [S][n_l][32]
  const float* W2[4];        // [S][32][32]
  const float* W3[4];        // [S][32][32]
  const float* W4[4];        // [S][32][n_l]
  float*       out;          // concat: [E,20][E,16][E,12][E,8]
  int          E;
};

__global__ __launch_bounds__(BLOCK) void radial_basis_wmma(Params p) {
  __shared__ _Float16 wlds[WLDS_HALVES];

  const int tid  = threadIdx.x;
  const int wave = tid >> 5;
  const int lane = tid & 31;
  const int hl   = lane >> 4;   // K-half selector
  const int col  = lane & 15;   // edge within this wave's 16-edge tile

  const int edge = blockIdx.x * EDGES_PER_BLOCK + wave * 16 + col;
  const int eC   = edge < p.E ? edge : p.E - 1;

  const float rv = p.r[eC];
  const int   sp = p.species[eC];

  // cubic spline: idx = clip(floor(r/dx), 0, 598), t = r - idx*dx
  const float dx = 5.0f / 599.0f;
  int idx = (int)floorf(rv / dx);
  idx = idx < 0 ? 0 : (idx > 598 ? 598 : idx);
  const float  t  = rv - (float)idx * dx;
  const float* cb = p.coeffs + (size_t)idx * 224;

  const int NLs[4]  = {20, 16, 12, 8};
  const int FOFF[4] = {0, 20, 36, 48};

  for (int l = 0; l < 4; ++l) {
    const int nl = NLs[l];
    const float* w1 = p.W1[l];
    const float* w2 = p.W2[l];
    const float* w3 = p.W3[l];
    const float* w4 = p.W4[l];

    __syncthreads();
    // Stage all 4 species' weights for this l into LDS, pre-swizzled into the
    // exact A-fragment order (A = W^T): lane lam, element i -> M = tile*16 +
    // (lam&15), K = i + (lam>>4)*16.  Zero-pad K>=n_l / M>=n_l.
    for (int j = tid; j < WLDS_HALVES; j += BLOCK) {
      const int i     = j & 15;
      const int lam   = (j >> 4) & 31;
      const int tile  = (j >> 9) & 1;
      const int layer = (j >> 10) & 3;
      const int s     = (j >> 12) & 3;
      const int M     = tile * 16 + (lam & 15);
      const int K     = i + (lam >> 4) * 16;
      float v = 0.0f;
      if (layer == 0)      { if (K < nl) v = w1[((size_t)s * nl + K) * 32 + M]; }
      else if (layer == 1) { v = w2[(size_t)s * 1024 + K * 32 + M]; }
      else if (layer == 2) { v = w3[(size_t)s * 1024 + K * 32 + M]; }
      else                 { if (M < nl) v = w4[((size_t)s * 32 + K) * nl + M]; }
      wlds[j] = (_Float16)v;
    }
    __syncthreads();

    // B = X^T fragment straight from the spline: lane holds K = e + 16*hl of
    // its edge's features for this l (zero beyond n_l).
    v16h bx;
#pragma unroll
    for (int e = 0; e < 16; ++e) {
      const int K = e + 16 * hl;
      float v = 0.0f;
      if (K < nl) {
        const int f = FOFF[l] + K;
        v = cb[f] + t * (cb[56 + f] + t * (cb[112 + f] + t * cb[168 + f]));
      }
      bx[e] = (_Float16)v;
    }

    float acc1[8], acc2[8];
#pragma unroll
    for (int rr = 0; rr < 8; ++rr) { acc1[rr] = 0.0f; acc2[rr] = 0.0f; }

    for (int s = 0; s < 4; ++s) {
      const _Float16* ws = wlds + ((size_t)s << 12);
#define FRAG(L, T) (*(const v16h*)(ws + (((L) * 2 + (T)) << 9) + (lane << 4)))

      v8f c10 = {}, c11 = {};
      c10 = wmma16x16x32(FRAG(0, 0), bx, c10);   // H1^T rows 0..15
      c11 = wmma16x16x32(FRAG(0, 1), bx, c11);   // H1^T rows 16..31
      v16h b2 = silu_transpose(c10, c11, hl);

      v8f c20 = {}, c21 = {};
      c20 = wmma16x16x32(FRAG(1, 0), b2, c20);
      c21 = wmma16x16x32(FRAG(1, 1), b2, c21);
      v16h b3 = silu_transpose(c20, c21, hl);

      v8f c30 = {}, c31 = {};
      c30 = wmma16x16x32(FRAG(2, 0), b3, c30);
      c31 = wmma16x16x32(FRAG(2, 1), b3, c31);
      v16h b4 = silu_transpose(c30, c31, hl);

      v8f y0 = {};
      y0 = wmma16x16x32(FRAG(3, 0), b4, y0);     // out dims 0..15
      const float m = (sp == s) ? 1.0f : 0.0f;
#pragma unroll
      for (int rr = 0; rr < 8; ++rr) acc1[rr] += m * y0[rr];
      if (nl > 16) {                              // only l=0 (n=20) needs tile 1
        v8f y1 = {};
        y1 = wmma16x16x32(FRAG(3, 1), b4, y1);   // out dims 16..31
#pragma unroll
        for (int rr = 0; rr < 8; ++rr) acc2[rr] += m * y1[rr];
      }
#undef FRAG
    }

    if (edge < p.E) {
      float* ob = p.out + (size_t)FOFF[l] * p.E + (size_t)edge * nl;
#pragma unroll
      for (int rr = 0; rr < 8; ++rr) {
        const int d = rr + 8 * hl;                // lane half owns dims 0..7 / 8..15
        if (d < nl) ob[d] = acc1[rr];
      }
      if (nl > 16) {
#pragma unroll
        for (int rr = 0; rr < 8; ++rr) {
          const int d = 16 + rr + 8 * hl;
          if (d < nl) ob[d] = acc2[rr];
        }
      }
    }
  }
}

extern "C" void kernel_launch(void* const* d_in, const int* in_sizes, int n_in,
                              void* d_out, int out_size, void* d_ws, size_t ws_size,
                              hipStream_t stream) {
  (void)n_in; (void)out_size; (void)d_ws; (void)ws_size;
  Params p;
  p.r       = (const float*)d_in[0];
  p.species = (const int*)d_in[1];
  p.coeffs  = (const float*)d_in[2];
  for (int l = 0; l < 4; ++l) {
    p.W1[l] = (const float*)d_in[3 + l];
    p.W2[l] = (const float*)d_in[7 + l];
    p.W3[l] = (const float*)d_in[11 + l];
    p.W4[l] = (const float*)d_in[15 + l];
  }
  p.out = (float*)d_out;
  p.E   = in_sizes[0];
  const int grid = (p.E + EDGES_PER_BLOCK - 1) / EDGES_PER_BLOCK;
  radial_basis_wmma<<<grid, BLOCK, 0, stream>>>(p);
}
